// DeformConv_82068235092075
// MI455X (gfx1250) — matrix operand order repaired and checked by reading, not compiled, not run on hardware
//
#include <hip/hip_runtime.h>
#include <hip/hip_bf16.h>
#include <cstdint>
#include <cstddef>

typedef __bf16 bf16_t;
typedef __attribute__((ext_vector_type(8)))  __bf16 v8bf;
typedef __attribute__((ext_vector_type(16))) __bf16 v16bf;
typedef __attribute__((ext_vector_type(8)))  float  v8f;
typedef unsigned int u32;
typedef __attribute__((ext_vector_type(4))) u32 u32x4;
typedef __attribute__((ext_vector_type(4))) int i32x4;
typedef __attribute__((ext_vector_type(8))) int i32x8;

#define HW    16384
#define CIN   256
#define OUTC  256
#define KTAP  9
#define NPIX  32768   // B * H * W
#define KC    2304    // KTAP * CIN

#if defined(__AMDGCN__) && __has_builtin(__builtin_amdgcn_tensor_load_to_lds)
#define TDM_OK 1
#else
#define TDM_OK 0
#endif

// ---------------------------------------------------------------------------
// Prep: x NCHW f32 -> NHWC bf16 (channel-contiguous for bilinear gathers)
// ---------------------------------------------------------------------------
__global__ __launch_bounds__(256) void k_prep_x(const float* __restrict__ x,
                                                bf16_t* __restrict__ xh) {
  const int idx = blockIdx.x * 256 + threadIdx.x;          // 8,388,608 elems
  const int b = idx >> 22;
  const int rem = idx & ((1 << 22) - 1);
  const int c = rem >> 14;
  const int hw = rem & (HW - 1);
  xh[(((size_t)b << 14) + hw) * CIN + c] = (bf16_t)x[idx];
}

// Weights [O][C][3][3] f32 -> wt2[o][k*256+c] bf16 (GEMM A-matrix, row-major)
__global__ __launch_bounds__(256) void k_prep_w(const float* __restrict__ w,
                                                bf16_t* __restrict__ wt2) {
  const int idx = blockIdx.x * 256 + threadIdx.x;          // 589,824 elems
  const int o = idx / KC;
  const int kc = idx % KC;
  const int k = kc >> 8, c = kc & 255;
  wt2[idx] = (bf16_t)w[(size_t)(o * CIN + c) * 9 + k];
}

// Offset weights [18][C][3][3] -> wofft[t][c][18]
__global__ __launch_bounds__(256) void k_prep_woff(const float* __restrict__ w_off,
                                                   float* __restrict__ wofft) {
  const int idx = blockIdx.x * 256 + threadIdx.x;
  if (idx >= 9 * 256 * 18) return;
  const int i = idx % 18;
  const int rc = idx / 18;
  const int t = rc >> 8;
  const int c = rc & 255;
  wofft[idx] = w_off[(size_t)(i * CIN + c) * 9 + t];
}

// ---------------------------------------------------------------------------
// Offset conv + bilinear tables: per (b, tap, pixel) -> 4 corner idx + 4 wts
// ---------------------------------------------------------------------------
__global__ __launch_bounds__(256) void k_offset(const float* __restrict__ x,
                                                const float* __restrict__ wofft,
                                                const float* __restrict__ b_off,
                                                int4* __restrict__ tidx,
                                                float4* __restrict__ twt) {
  const int P = blockIdx.x * 256 + threadIdx.x;            // 32768 pixels
  const int b = P >> 14, hw = P & (HW - 1);
  const int h = hw >> 7, w = hw & 127;

  float acc[18];
#pragma unroll
  for (int i = 0; i < 18; ++i) acc[i] = b_off[i];

  for (int t = 0; t < 9; ++t) {
    const int yy = h + t / 3 - 1, xx = w + t % 3 - 1;
    if (yy < 0 || yy >= 128 || xx < 0 || xx >= 128) continue;
    const float* xp = x + (size_t)b * CIN * HW + yy * 128 + xx;
    const float* wp = wofft + (size_t)t * 256 * 18;
    for (int c = 0; c < 256; ++c) {
      const float xv = xp[(size_t)c * HW];
      const float* wv = wp + c * 18;
#pragma unroll
      for (int i = 0; i < 18; ++i) acc[i] += xv * wv[i];
    }
  }

#pragma unroll
  for (int k = 0; k < 9; ++k) {
    const float py = (float)h + (float)(k / 3 - 1) + acc[2 * k];
    const float px = (float)w + (float)(k % 3 - 1) + acc[2 * k + 1];
    const float y0f = floorf(py), x0f = floorf(px);
    const float ly = py - y0f, lx = px - x0f;
    const int y0 = (int)y0f, x0 = (int)x0f;
    const int ys[2] = {y0, y0 + 1};
    const int xs[2] = {x0, x0 + 1};
    const float wy[2] = {1.f - ly, ly};
    const float wx[2] = {1.f - lx, lx};
    int ic[4]; float wc[4];
#pragma unroll
    for (int j = 0; j < 4; ++j) {
      const int yc = ys[j >> 1], xc = xs[j & 1];
      const bool ok = (yc >= 0) && (yc < 128) && (xc >= 0) && (xc < 128);
      ic[j] = ok ? (yc * 128 + xc) : 0;
      wc[j] = ok ? (wy[j >> 1] * wx[j & 1]) : 0.f;
    }
    const int o = ((b * KTAP + k) << 14) + hw;
    tidx[o] = make_int4(ic[0], ic[1], ic[2], ic[3]);
    twt[o]  = make_float4(wc[0], wc[1], wc[2], wc[3]);
  }
}

// ---------------------------------------------------------------------------
// TDM descriptor: 2-D tile (32 contiguous kc x 256 o-rows, row stride KC)
// from wt2 into LDS, row-major [o][32]. See CDNA5 ISA.
// ---------------------------------------------------------------------------
#if TDM_OK
__device__ __forceinline__ void tdm_load_wtile(const bf16_t* gsrc, u32 lds_off) {
  const unsigned long long ga = (unsigned long long)(uintptr_t)gsrc;
  u32x4 g0;
  g0.x = 1u;                                             // count=1, user mode
  g0.y = lds_off;                                        // lds_addr
  g0.z = (u32)ga;                                        // global_addr[31:0]
  g0.w = (u32)((ga >> 32) & 0x01FFFFFFu) | (2u << 30);   // addr[56:32] | type=2
  i32x8 g1;
  g1[0] = (int)(1u << 16);      // workgroup_mask=0, data_size=1 (2 bytes)
  g1[1] = (int)(2304u << 16);   // tensor_dim0 = 2304 (row length, OOB check)
  g1[2] = (int)(256u << 16);    // tensor_dim1 = 256 rows
  g1[3] = (int)(32u << 16);     // tile_dim0 = 32 (contiguous kc)
  g1[4] = (int)256;             // tile_dim1 = 256, tile_dim2 = 0
  g1[5] = (int)2304;            // tensor_dim0_stride = 2304 (low 32)
  g1[6] = 0;                    // stride hi / dim1_stride (unused, 2-D)
  g1[7] = 0;
  const i32x4 z4 = {0, 0, 0, 0};
#if defined(__clang_major__) && __clang_major__ >= 23
  const i32x8 z8 = {0, 0, 0, 0, 0, 0, 0, 0};
  __builtin_amdgcn_tensor_load_to_lds(g0, g1, z4, z4, z8, 0);
#else
  __builtin_amdgcn_tensor_load_to_lds(g0, g1, z4, z4, 0);
#endif
}
#endif

// ---------------------------------------------------------------------------
// Fused deformable-conv GEMM: out_raw[o][pixel] = sum_{k,c} w * val
// A = weights (16 o x 32 kc), B = bilinear-sampled values (32 kc x 16 px).
// Block: 256 o-channels x 64 pixels, 8 waves, 8 WMMA/chunk/wave, 72 chunks.
// Weight tiles are DMA'd by the Tensor Data Mover with ping-pong buffering.
// ---------------------------------------------------------------------------
__global__ __launch_bounds__(256) void k_deform_gemm(const bf16_t* __restrict__ xh,
                                                     const bf16_t* __restrict__ wt2,
                                                     const int4* __restrict__ tidx,
                                                     const float4* __restrict__ twt,
                                                     float* __restrict__ out_raw) {
  __shared__ bf16_t sV[64 * 32];       // sampled values  [pixel][kc-chunk]
  __shared__ bf16_t sW[2][256 * 32];   // weight tiles    [o][kc-chunk], ping-pong

  const int tid = threadIdx.x;
  const int lane = tid & 31;
  const int wave = tid >> 5;
  const int P0 = blockIdx.x * 64;            // 64 pixels, within one image
  const int b = P0 >> 14;
  const int hw0 = P0 & (HW - 1);
  const int p = tid >> 2;                    // pixel within tile (0..63)
  const int cg = (tid & 3) * 8;              // channel subgroup (0,8,16,24)
  const bf16_t* xb = xh + (size_t)b * HW * CIN;

  v8f acc[8] = {};

  // A-fragment addressing (16-bit A 16x32): wave owns o-tiles 2w, 2w+1
  const int arow = (wave * 2) * 16 + (lane & 15);
  const int akb = (lane >> 4) * 8;
  // B-fragment addressing (16-bit B 32x16): lane holds pixel column
  const int bn = lane & 15;
  const int bkb = (lane >> 4) * 8;

#if TDM_OK
  if (wave == 0) tdm_load_wtile(wt2, (u32)(uintptr_t)&sW[0][0]);
#endif

  int4 ci = {};
  float4 cw = {};

  for (int c = 0; c < 72; ++c) {
    const int k = c >> 3;
    const int buf = c & 1;
    if ((c & 7) == 0) {
      const int tb = ((b * KTAP + k) << 14) + hw0 + p;
      ci = tidx[tb];
      cw = twt[tb];
      if (k + 1 < KTAP) {
        __builtin_prefetch((const void*)(tidx + tb + (1 << 14)), 0, 0);
        __builtin_prefetch((const void*)(twt + tb + (1 << 14)), 0, 0);
      }
    }
    // ---- build value tile: bilinear gather of 64 px x 32 kc ----
    const int c0 = (c & 7) * 32 + cg;
    const v8bf l00 = *(const v8bf*)(xb + (size_t)ci.x * CIN + c0);
    const v8bf l01 = *(const v8bf*)(xb + (size_t)ci.y * CIN + c0);
    const v8bf l10 = *(const v8bf*)(xb + (size_t)ci.z * CIN + c0);
    const v8bf l11 = *(const v8bf*)(xb + (size_t)ci.w * CIN + c0);
    v8bf av;
#pragma unroll
    for (int i = 0; i < 8; ++i) {
      const float v = cw.x * (float)l00[i] + cw.y * (float)l01[i] +
                      cw.z * (float)l10[i] + cw.w * (float)l11[i];
      av[i] = (bf16_t)v;
    }
    *(v8bf*)(&sV[p * 32 + cg]) = av;

    // ---- stage next weight tile (TDM ping-pong) / sync fallback ----
#if TDM_OK
    if (wave == 0) {
      if (c + 1 < 72) {
        tdm_load_wtile(wt2 + (size_t)(c + 1) * 32, (u32)(uintptr_t)&sW[(c + 1) & 1][0]);
        __builtin_amdgcn_s_wait_tensorcnt(1);   // current tile complete
      } else {
        __builtin_amdgcn_s_wait_tensorcnt(0);
      }
    }
#else
    {
      const v8bf* s = (const v8bf*)(wt2 + (size_t)tid * KC + (size_t)c * 32);
      v8bf* d = (v8bf*)&sW[buf][tid * 32];
      d[0] = s[0]; d[1] = s[1]; d[2] = s[2]; d[3] = s[3];
    }
#endif
    __syncthreads();

    // ---- WMMA: 2 A-tiles x 4 B-tiles per wave ----
    const bf16_t* sWb = sW[buf];
    v16bf afrag[2];
#pragma unroll
    for (int wi = 0; wi < 2; ++wi) {
      const v8bf alo = *(const v8bf*)(&sWb[(arow + wi * 16) * 32 + akb]);
      const v8bf ahi = *(const v8bf*)(&sWb[(arow + wi * 16) * 32 + akb + 16]);
      afrag[wi] = __builtin_shufflevector(
          alo, ahi, 0, 1, 2, 3, 4, 5, 6, 7, 8, 9, 10, 11, 12, 13, 14, 15);
    }
#pragma unroll
    for (int nj = 0; nj < 4; ++nj) {
      const v8bf blo = *(const v8bf*)(&sV[(nj * 16 + bn) * 32 + bkb]);
      const v8bf bhi = *(const v8bf*)(&sV[(nj * 16 + bn) * 32 + bkb + 16]);
      const v16bf bfrag = __builtin_shufflevector(
          blo, bhi, 0, 1, 2, 3, 4, 5, 6, 7, 8, 9, 10, 11, 12, 13, 14, 15);
#pragma unroll
      for (int wi = 0; wi < 2; ++wi) {
        acc[wi * 4 + nj] = __builtin_amdgcn_wmma_f32_16x16x32_bf16(
            false, afrag[wi], false, bfrag, (short)0, acc[wi * 4 + nj], false, false);
      }
    }
    __syncthreads();
  }

  // ---- store D tiles: out_raw[o][pixel], coalesced along pixels ----
  const int vrow = (lane >> 4) << 3;
#pragma unroll
  for (int wi = 0; wi < 2; ++wi) {
#pragma unroll
    for (int nj = 0; nj < 4; ++nj) {
      const int pix = P0 + nj * 16 + bn;
#pragma unroll
      for (int v = 0; v < 8; ++v) {
        const int o = (wave * 2 + wi) * 16 + vrow + v;
        out_raw[(size_t)o * NPIX + pix] = acc[wi * 4 + nj][v];
      }
    }
  }
}

// ---------------------------------------------------------------------------
// BN batch stats: one block per channel, contiguous reduction, no atomics
// ---------------------------------------------------------------------------
__global__ __launch_bounds__(256) void k_stats(const float* __restrict__ oraw,
                                               float* __restrict__ stats) {
  __shared__ float r1[256], r2[256];
  const int o = blockIdx.x;
  const float* src = oraw + (size_t)o * NPIX;
  float s = 0.f, s2 = 0.f;
  for (int p = threadIdx.x; p < NPIX; p += 256) {
    const float v = src[p];
    s += v;
    s2 += v * v;
  }
  r1[threadIdx.x] = s;
  r2[threadIdx.x] = s2;
  __syncthreads();
  for (int st = 128; st > 0; st >>= 1) {
    if (threadIdx.x < st) {
      r1[threadIdx.x] += r1[threadIdx.x + st];
      r2[threadIdx.x] += r2[threadIdx.x + st];
    }
    __syncthreads();
  }
  if (threadIdx.x == 0) {
    stats[o] = r1[0];
    stats[OUTC + o] = r2[0];
  }
}

// ---------------------------------------------------------------------------
// BN + ReLU, [o][pixel] -> NCHW f32 (both sides coalesced, no transpose)
// ---------------------------------------------------------------------------
__global__ __launch_bounds__(256) void k_final(const float* __restrict__ oraw,
                                               const float* __restrict__ stats,
                                               const float* __restrict__ gamma,
                                               const float* __restrict__ beta,
                                               float* __restrict__ out) {
  const int idx = blockIdx.x * 256 + threadIdx.x;   // 8,388,608
  const int o = idx >> 15;
  const int p = idx & (NPIX - 1);
  const float mean = stats[o] * (1.f / 32768.f);
  const float var = stats[OUTC + o] * (1.f / 32768.f) - mean * mean;
  const float sc = gamma[o] * rsqrtf(var + 1e-5f);
  const float sh = beta[o] - mean * sc;
  const float v = oraw[(size_t)o * NPIX + p] * sc + sh;
  const int b = p >> 14, hw = p & (HW - 1);
  out[(((size_t)b * OUTC + o) << 14) + hw] = fmaxf(v, 0.f);
}

// ---------------------------------------------------------------------------
extern "C" void kernel_launch(void* const* d_in, const int* in_sizes, int n_in,
                              void* d_out, int out_size, void* d_ws, size_t ws_size,
                              hipStream_t stream) {
  const float* x     = (const float*)d_in[0];
  const float* w_off = (const float*)d_in[1];
  const float* b_off = (const float*)d_in[2];
  const float* w     = (const float*)d_in[3];
  const float* gamma = (const float*)d_in[4];
  const float* beta  = (const float*)d_in[5];
  float* out = (float*)d_out;

  char* ws = (char*)d_ws;
  bf16_t* xh    = (bf16_t*)(ws + 0);          // 16,777,216 B
  bf16_t* wt2   = (bf16_t*)(ws + 16777216);   //  1,179,648 B
  float*  wofft = (float*)(ws + 17956864);    //    165,888 B
  int4*   tidx  = (int4*)(ws + 18122752);     //  4,718,592 B
  float4* twt   = (float4*)(ws + 22841344);   //  4,718,592 B
  float*  oraw  = (float*)(ws + 27559936);    // 33,554,432 B
  float*  stats = (float*)(ws + 61114368);    //      2,048 B

  k_prep_x<<<32768, 256, 0, stream>>>(x, xh);
  k_prep_w<<<2304, 256, 0, stream>>>(w, wt2);
  k_prep_woff<<<162, 256, 0, stream>>>(w_off, wofft);
  k_offset<<<128, 256, 0, stream>>>(x, wofft, b_off, tidx, twt);
  k_deform_gemm<<<512, 256, 0, stream>>>(xh, wt2, tidx, twt, oraw);
  k_stats<<<256, 256, 0, stream>>>(oraw, stats);
  k_final<<<32768, 256, 0, stream>>>(oraw, stats, gamma, beta, out);
}
